// PCD_Align_65000035058383
// MI455X (gfx1250) — matrix operand order repaired and checked by reading, not compiled, not run on hardware
//
#include <hip/hip_runtime.h>
#include <hip/hip_bf16.h>

// ---------------------------------------------------------------------------
// PCD-align for MI455X (gfx1250, wave32).
// Convs = implicit GEMM on v_wmma_f32_16x16x32_bf16 (bf16 in, fp32 acc).
//   K is packed TAP-MAJOR: k = tap*Cin + c  (Cin % 32 == 0 everywhere), so a
//   32-K chunk has one fixed tap and 32 consecutive channels -> the 32-pixel
//   B tile is staged into LDS with contiguous 16B global loads (stride-1 and
//   1x1 modes). Weights are repacked on-device to [Cout][tap*Cin+c] so the A
//   fragment is always 2x global_load_b128. B fragments read back from the
//   [pixel][k] LDS tile as 2x ds_load_b128. Double-buffered, 1 barrier/chunk.
//   Block = 32 * ceil(Cout/16) threads (6 or 14 waves); wave = one 16-channel
//   M tile over NT=2 pixel subtiles (2 WMMAs per chunk per wave).
// Deformable sampling is VALU bilinear; its 9-tap reduction is the K=864
// WMMA GEMM (1x1 mode) with tap-major sampled layout.
// ---------------------------------------------------------------------------

typedef __attribute__((ext_vector_type(16))) __bf16 v16bf;
typedef __attribute__((ext_vector_type(8)))  __bf16 v8bf;
typedef __attribute__((ext_vector_type(8)))  float  v8f;

#define NT 2                      // pixel subtiles (16 px each) per block

static __device__ __forceinline__ __bf16 f2bf(float f) {
  unsigned u = __builtin_bit_cast(unsigned, f);
  unsigned r = (u + 0x7FFFu + ((u >> 16) & 1u)) >> 16;   // round-to-nearest-even
  unsigned short s = (unsigned short)r;
  return __builtin_bit_cast(__bf16, s);
}

// ------------------------- elementwise helpers -----------------------------

__global__ void cvt_f32_bf16_k(const float* __restrict__ src,
                               __bf16* __restrict__ dst, long n) {
  long i = (long)blockIdx.x * blockDim.x + threadIdx.x;
  if (i < n) dst[i] = f2bf(src[i]);
}

// weights -> bf16 [Cout][tap*Cin + c].
// transp=0: src (Cout,Cin,3,3).  transp=1: src (Cin,Cout,3,3) (deconv).
__global__ void repack_w_k(const float* __restrict__ w, __bf16* __restrict__ dst,
                           int Cin, int Cout, int transp) {
  long n = (long)Cin * Cout * 9;
  long i = (long)blockIdx.x * blockDim.x + threadIdx.x;
  if (i >= n) return;
  const int K = Cin * 9;
  int o = (int)(i / K);
  int r = (int)(i % K);
  int t = r / Cin;
  int c = r % Cin;
  float v = transp ? w[((long)c * Cout + o) * 9 + t]
                   : w[((long)o * Cin + c) * 9 + t];
  dst[i] = f2bf(v);
}

__global__ void round_q_k(const float* __restrict__ en, float* __restrict__ q,
                          __bf16* __restrict__ qbf, long n) {
  long i = (long)blockIdx.x * blockDim.x + threadIdx.x;
  if (i < n) {
    float v = rintf(en[i]);          // jnp.round == round-half-to-even
    q[i] = v;
    qbf[i] = f2bf(v);
  }
}

__global__ void add_inplace_k(float* __restrict__ a, const float* __restrict__ b,
                              long n) {
  long i = (long)blockIdx.x * blockDim.x + threadIdx.x;
  if (i < n) a[i] = a[i] + b[i];
}

// ------------------------- DCNv1 bilinear sampling -------------------------
// smp[(tap*C + c)*HW + p] = bilinear(xref[c], y+dy+off_y, x+dx+off_x), 0-pad.
__global__ void deform_sample_k(const float* __restrict__ xref,  // [C,H,W] (one batch)
                                const float* __restrict__ off,   // [2*G*9,H,W]
                                __bf16* __restrict__ smp,        // [9*C,H,W] tap-major
                                int H, int W, int C) {
  const int HW = H * W;
  const long total = (long)C * 9 * HW;
  long tid = (long)blockIdx.x * blockDim.x + threadIdx.x;
  if (tid >= total) return;
  int p   = (int)(tid % HW);
  int r   = (int)(tid / HW);
  int c   = r % C;
  int tap = r / C;
  int y = p / W, x = p % W;
  int g = c >> 3;                                  // Cg = 8 channels per group
  int chy = (g * 9 + tap) * 2;                     // (dy,dx) channel pair
  float py = (float)y + (float)(tap / 3 - 1) + off[(long)chy * HW + p];
  float px = (float)x + (float)(tap % 3 - 1) + off[(long)(chy + 1) * HW + p];
  float y0 = floorf(py), x0 = floorf(px);
  float wy = py - y0,    wx = px - x0;
  int y0i = (int)y0, x0i = (int)x0;
  const float* xc = xref + (long)c * HW;
  float acc = 0.f;
#pragma unroll
  for (int oy = 0; oy < 2; ++oy)
#pragma unroll
    for (int ox = 0; ox < 2; ++ox) {
      int yy = y0i + oy, xx = x0i + ox;
      if (yy >= 0 && yy < H && xx >= 0 && xx < W) {
        float wgt = (oy ? wy : 1.f - wy) * (ox ? wx : 1.f - wx);
        acc += wgt * xc[yy * W + xx];
      }
    }
  smp[tid] = f2bf(acc);
}

// ------------------------- WMMA implicit-GEMM conv -------------------------
// MODE 0: 3x3 stride 1, SAME (pad_beg = 1)      -> vectorized staging
// MODE 1: 3x3 stride 2, SAME (pad_beg = 0)      -> scalar staging (small)
// MODE 2: 3x3 stride-2 conv-transpose (oy=2*iy+dy-1) -> scalar staging (small)
// MODE 3: 1x1 (DCN gemm), K = Cin               -> vectorized staging
template <int MODE>
__global__ __launch_bounds__(448)
void conv_wmma_k(const __bf16* __restrict__ xA, const __bf16* __restrict__ xB,
                 int CinA,
                 const __bf16* __restrict__ wbf, const float* __restrict__ bias,
                 float* __restrict__ outF, __bf16* __restrict__ outBF,
                 int Cin, int Cout, int inH, int inW, int outH, int outW,
                 int do_lrelu) {
  alignas(16) __shared__ __bf16 sB[2][NT * 16 * 32];   // [pixel][k], double buffer
  const int tid    = threadIdx.x;
  const int lane   = tid & 31;
  const int wave   = tid >> 5;
  const int nthr   = blockDim.x;
  const int tilesX = outW / (NT * 16);
  const int oy     = blockIdx.x / tilesX;
  const int px0    = (blockIdx.x % tilesX) * (NT * 16);
  const int b      = blockIdx.z;
  const int inHW   = inH * inW;
  const int K      = (MODE == 3) ? Cin : Cin * 9;
  const int CinB   = Cin - CinA;
  const __bf16* baseA = xA + (long)b * CinA * inHW;
  const __bf16* baseB = xB ? (xB + (long)b * CinB * inHW) : (const __bf16*)0;

  const __bf16 bz = __builtin_bit_cast(__bf16, (unsigned short)0);
  const int m     = wave * 16 + (lane & 15);
  const bool mval = (m < Cout);
  const int kAoff = (lane >> 4) << 3;   // A: lanes 16-31 hold K+8
  const int kBoff = (lane >> 4) << 3;   // B: lanes 16-31 hold K+8

  v8f acc[NT];
#pragma unroll
  for (int nt = 0; nt < NT; ++nt)
    acc[nt] = (v8f){0.f, 0.f, 0.f, 0.f, 0.f, 0.f, 0.f, 0.f};

  const int nkb = K >> 5;
  for (int ib = 0; ib < nkb; ++ib) {
    const int kb = ib << 5;
    __bf16* sbuf = sB[ib & 1];

    if (MODE == 0 || MODE == 3) {
      // ---- vectorized staging: tap fixed per chunk, channels contiguous ----
      // unit u: kk = u/(2*NT) (0..31), 8-pixel group = u%(2*NT)
      const int t0 = (MODE == 3) ? 0 : kb / Cin;
      const int c0 = (MODE == 3) ? kb : kb % Cin;
      const int dy = t0 / 3, dx = t0 % 3;
      const int iy = (MODE == 3) ? oy : (oy + dy - 1);
      const bool rowok = (iy >= 0) && (iy < inH);
      for (int u = tid; u < 32 * 2 * NT; u += nthr) {
        const int kk  = u / (2 * NT);
        const int grp = u % (2 * NT);
        const int ix0 = (MODE == 3) ? (px0 + grp * 8) : (px0 + grp * 8 + dx - 1);
        const int c   = c0 + kk;
        const __bf16* bp = baseA;
        int cc = c;
        if (baseB && c >= CinA) { bp = baseB; cc = c - CinA; }
        const __bf16* src = bp + (long)cc * inHW + (long)iy * inW + ix0;
        __bf16 vals[8];
        if (rowok && ix0 >= 0 && (ix0 + 8) <= inW) {
          __builtin_memcpy(vals, src, 16);        // contiguous 8x bf16
        } else {
#pragma unroll
          for (int e = 0; e < 8; ++e) {
            const int ix = ix0 + e;
            vals[e] = (rowok && ix >= 0 && ix < inW) ? src[e] : bz;
          }
        }
#pragma unroll
        for (int e = 0; e < 8; ++e)
          sbuf[(grp * 8 + e) * 32 + kk] = vals[e];
      }
    } else {
      // ---- scalar staging (stride-2 conv / deconv; small layers) ----
      for (int idx = tid; idx < NT * 16 * 32; idx += nthr) {
        const int pe = idx >> 5;
        const int kk = idx & 31;
        const int k  = kb + kk;
        const int t  = k / Cin, c = k % Cin;     // tap-major decode
        const int dy = t / 3, dx = t % 3;
        const int ppx = px0 + pe;
        int iy, ix;
        bool ok = true;
        if (MODE == 1) { iy = oy * 2 + dy; ix = ppx * 2 + dx; }
        else {           // MODE 2 (deconv): oy = 2*iy + dy - 1
          const int ny = oy + 1 - dy, nx = ppx + 1 - dx;
          ok = ((ny & 1) == 0) && ((nx & 1) == 0) && ny >= 0 && nx >= 0;
          iy = ny >> 1; ix = nx >> 1;
        }
        __bf16 v = bz;
        if (ok && iy >= 0 && iy < inH && ix >= 0 && ix < inW) {
          const __bf16* bp = baseA;
          int cc = c;
          if (baseB && c >= CinA) { bp = baseB; cc = c - CinA; }
          v = bp[(long)cc * inHW + (long)iy * inW + ix];
        }
        sbuf[pe * 32 + kk] = v;
      }
    }
    __syncthreads();

    // ---- A fragment: 2x global_load_b128 from packed [Cout][K] weights ----
    v16bf A;
    if (mval) {
      const __bf16* ap = wbf + (long)m * K + kb + kAoff;
      const v8bf lo = *(const v8bf*)ap;
      const v8bf hi = *(const v8bf*)(ap + 16);
#pragma unroll
      for (int e = 0; e < 8; ++e) { A[e] = lo[e]; A[e + 8] = hi[e]; }
    } else {
#pragma unroll
      for (int e = 0; e < 16; ++e) A[e] = bz;
    }
    // ---- B fragments (one per pixel subtile): 2x ds_load_b128 each ----
#pragma unroll
    for (int nt = 0; nt < NT; ++nt) {
      const __bf16* bp = &sbuf[((lane & 15) + nt * 16) * 32 + kBoff];
      const v8bf blo = *(const v8bf*)bp;
      const v8bf bhi = *(const v8bf*)(bp + 16);
      v16bf Bf;
#pragma unroll
      for (int e = 0; e < 8; ++e) { Bf[e] = blo[e]; Bf[e + 8] = bhi[e]; }
      acc[nt] = __builtin_amdgcn_wmma_f32_16x16x32_bf16(
          false, A, false, Bf, (short)0, acc[nt], false, false);
    }
    // double-buffered: the one barrier above also fences the next overwrite
  }

  // ---- D layout: lane%16 = pixel N; VGPR v -> M = v + (lane/16)*8 ----
  const long outHW = (long)outH * outW;
#pragma unroll
  for (int nt = 0; nt < NT; ++nt) {
    const int px = px0 + nt * 16 + (lane & 15);
#pragma unroll
    for (int v = 0; v < 8; ++v) {
      const int oc = wave * 16 + v + ((lane >> 4) << 3);
      if (oc < Cout) {
        float r = acc[nt][v] + bias[oc];
        if (do_lrelu) r = (r >= 0.f) ? r : 0.1f * r;
        const long oidx = ((long)b * Cout + oc) * outHW + (long)oy * outW + px;
        if (outF)  outF[oidx]  = r;
        if (outBF) outBF[oidx] = f2bf(r);
      }
    }
  }
}

// ---------------------------------------------------------------------------

extern "C" void kernel_launch(void* const* d_in, const int* in_sizes, int n_in,
                              void* d_out, int out_size, void* d_ws, size_t ws_size,
                              hipStream_t stream) {
  (void)in_sizes; (void)n_in; (void)out_size; (void)ws_size;
  const int B = 4, NF = 96, H = 128, W = 128;
  const int HW = H * W;                            // 16384
  const int H2 = 64, W2 = 64, HW2 = H2 * W2;       // 4096
  const int H4 = 32, W4 = 32, HW4 = H4 * W4;       // 1024
  const int G = 12, OFFC = 2 * G * 9;              // 216
  const int CG = NF * 9;                           // 864 (dcn gemm K)

  const float* ref   = (const float*)d_in[0];
  const float* inp   = (const float*)d_in[1];
  const float* w_oc1 = (const float*)d_in[2];  const float* b_oc1 = (const float*)d_in[3];
  const float* w_oc3 = (const float*)d_in[4];  const float* b_oc3 = (const float*)d_in[5];
  const float* w_e1  = (const float*)d_in[6];  const float* b_e1  = (const float*)d_in[7];
  const float* w_e2  = (const float*)d_in[8];  const float* b_e2  = (const float*)d_in[9];
  const float* w_d1  = (const float*)d_in[10]; const float* b_d1  = (const float*)d_in[11];
  const float* w_d2  = (const float*)d_in[12]; const float* b_d2  = (const float*)d_in[13];
  const float* w_off = (const float*)d_in[14]; const float* b_off = (const float*)d_in[15];
  const float* w_dcn = (const float*)d_in[16]; const float* b_dcn = (const float*)d_in[17];
  const float* w_c1  = (const float*)d_in[18]; const float* b_c1  = (const float*)d_in[19];
  const float* w_c2  = (const float*)d_in[20]; const float* b_c2  = (const float*)d_in[21];

  float* out    = (float*)d_out;
  float* outL1  = out;                                    // [4,96,128,128]
  float* outEn  = outL1 + (long)B * NF * HW;              // [4,96,32,32]
  float* outQ   = outEn + (long)B * NF * HW4;
  float* outDec = outQ  + (long)B * NF * HW4;             // [4,96,128,128]

  // ---- workspace bump allocator (256B aligned) ----
  char* wp = (char*)d_ws;
  auto alloc = [&](size_t bytes) -> void* {
    void* r = (void*)wp;
    wp += (bytes + 255) & ~(size_t)255;
    return r;
  };
  const size_t EB = sizeof(__bf16);
  // bf16 weights, packed [Cout][tap*Cin+c] on device
  __bf16* wb_oc1 = (__bf16*)alloc(EB * (size_t)NF * 2 * NF * 9);
  __bf16* wb_oc3 = (__bf16*)alloc(EB * (size_t)NF * NF * 9);
  __bf16* wb_e1  = (__bf16*)alloc(EB * (size_t)NF * NF * 9);
  __bf16* wb_e2  = (__bf16*)alloc(EB * (size_t)NF * NF * 9);
  __bf16* wb_d1  = (__bf16*)alloc(EB * (size_t)NF * NF * 9);
  __bf16* wb_d2  = (__bf16*)alloc(EB * (size_t)NF * NF * 9);
  __bf16* wb_off = (__bf16*)alloc(EB * (size_t)OFFC * NF * 9);
  __bf16* wb_dcn = (__bf16*)alloc(EB * (size_t)NF * NF * 9);
  __bf16* wb_c1  = (__bf16*)alloc(EB * (size_t)NF * 2 * NF * 9);
  __bf16* wb_c2  = (__bf16*)alloc(EB * (size_t)NF * NF * 9);
  // bf16 activations (with reuse)
  __bf16* refb = (__bf16*)alloc(EB * (size_t)B * NF * HW);
  __bf16* inpb = (__bf16*)alloc(EB * (size_t)B * NF * HW);   // reused as L1_bf
  __bf16* t1b  = (__bf16*)alloc(EB * (size_t)B * NF * HW);   // reused as dec_bf
  __bf16* tb   = (__bf16*)alloc(EB * (size_t)B * NF * HW);   // reused as o1_bf
  __bf16* e1b  = (__bf16*)alloc(EB * (size_t)B * NF * HW2);
  __bf16* qb   = (__bf16*)alloc(EB * (size_t)B * NF * HW4);
  __bf16* d1b  = (__bf16*)alloc(EB * (size_t)B * NF * HW2);
  __bf16* smpb = (__bf16*)alloc(EB * (size_t)CG * HW);       // one batch, tap-major
  // fp32 scratch
  float* off_f = (float*)alloc(sizeof(float) * (size_t)B * OFFC * HW);  // reused as o2_f
  __bf16* L1b  = inpb;   // reuse
  __bf16* decb = t1b;    // reuse
  __bf16* o1b  = tb;     // reuse
  float*  o2f  = off_f;  // reuse (after sampling is done)

  auto cvt = [&](const float* s, __bf16* d, long n) {
    cvt_f32_bf16_k<<<dim3((unsigned)((n + 255) / 256)), dim3(256), 0, stream>>>(s, d, n);
  };
  auto repack = [&](const float* s, __bf16* d, int Cin, int Cout, int transp) {
    long n = (long)Cin * Cout * 9;
    repack_w_k<<<dim3((unsigned)((n + 255) / 256)), dim3(256), 0, stream>>>(s, d, Cin, Cout, transp);
  };
  // ---- repack weights (tap-major K) + convert inputs to bf16 ----
  repack(w_oc1, wb_oc1, 2 * NF, NF, 0);
  repack(w_oc3, wb_oc3, NF, NF, 0);
  repack(w_e1,  wb_e1,  NF, NF, 0);
  repack(w_e2,  wb_e2,  NF, NF, 0);
  repack(w_d1,  wb_d1,  NF, NF, 1);      // deconv layout (Cin,Cout,3,3)
  repack(w_d2,  wb_d2,  NF, NF, 1);
  repack(w_off, wb_off, NF, OFFC, 0);
  repack(w_dcn, wb_dcn, NF, NF, 0);      // K=864: tap-major == [o][t*96+c]
  repack(w_c1,  wb_c1,  2 * NF, NF, 0);
  repack(w_c2,  wb_c2,  NF, NF, 0);
  cvt(ref, refb, (long)B * NF * HW);
  cvt(inp, inpb, (long)B * NF * HW);

  const dim3 blk6(32 * (NF / 16));                 // 192 threads, 6 m-tiles
  const dim3 blkOff(32 * ((OFFC + 15) / 16));      // 448 threads, 14 m-tiles
  const dim3 g128((W / (NT * 16)) * H, 1, B);      // 512 pixel tiles (32 px each)
  const dim3 g64((W2 / (NT * 16)) * H2, 1, B);     // 128
  const dim3 g32((W4 / (NT * 16)) * H4, 1, B);     //  32

  // t = lrelu(conv(cat(ref,inp), oc1)); t = lrelu(conv(t, oc3))
  conv_wmma_k<0><<<g128, blk6, 0, stream>>>(refb, inpb, NF, wb_oc1, b_oc1,
      nullptr, t1b, 2 * NF, NF, H, W, H, W, 1);
  conv_wmma_k<0><<<g128, blk6, 0, stream>>>(t1b, nullptr, NF, wb_oc3, b_oc3,
      nullptr, tb, NF, NF, H, W, H, W, 1);
  // encoder: e1 (s2, lrelu), e2 (s2) -> en (fp32 output)
  conv_wmma_k<1><<<g64, blk6, 0, stream>>>(tb, nullptr, NF, wb_e1, b_e1,
      nullptr, e1b, NF, NF, H, W, H2, W2, 1);
  conv_wmma_k<1><<<g32, blk6, 0, stream>>>(e1b, nullptr, NF, wb_e2, b_e2,
      outEn, nullptr, NF, NF, H2, W2, H4, W4, 0);
  // q = round(en)
  {
    long n = (long)B * NF * HW4;
    round_q_k<<<dim3((unsigned)((n + 255) / 256)), dim3(256), 0, stream>>>(outEn, outQ, qb, n);
  }
  // decoder: d1 (deconv, lrelu), d2 (deconv) -> dec (fp32 output + bf16)
  conv_wmma_k<2><<<g64, blk6, 0, stream>>>(qb, nullptr, NF, wb_d1, b_d1,
      nullptr, d1b, NF, NF, H4, W4, H2, W2, 1);
  conv_wmma_k<2><<<g128, blk6, 0, stream>>>(d1b, nullptr, NF, wb_d2, b_d2,
      outDec, decb, NF, NF, H2, W2, H, W, 0);
  // offset conv (Cout=216, fp32 only — feeds bilinear coordinates)
  conv_wmma_k<0><<<g128, blkOff, 0, stream>>>(decb, nullptr, NF, wb_off, b_off,
      off_f, nullptr, NF, OFFC, H, W, H, W, 0);
  // DCNv1: per batch, bilinear sample then K=864 WMMA gemm -> L1
  for (int b = 0; b < B; ++b) {
    long n = (long)CG * HW;
    deform_sample_k<<<dim3((unsigned)((n + 255) / 256)), dim3(256), 0, stream>>>(
        ref + (long)b * NF * HW, off_f + (long)b * OFFC * HW, smpb, H, W, NF);
    conv_wmma_k<3><<<dim3((W / (NT * 16)) * H, 1, 1), blk6, 0, stream>>>(
        smpb, nullptr, CG, wb_dcn, b_dcn,
        outL1 + (long)b * NF * HW, L1b + (long)b * NF * HW,
        CG, NF, H, W, H, W, 0);
  }
  // o = lrelu(conv(cat(L1, ref), c1)); o = lrelu(conv(o, c2)); L1 += o
  conv_wmma_k<0><<<g128, blk6, 0, stream>>>(L1b, refb, NF, wb_c1, b_c1,
      nullptr, o1b, 2 * NF, NF, H, W, H, W, 1);
  conv_wmma_k<0><<<g128, blk6, 0, stream>>>(o1b, nullptr, NF, wb_c2, b_c2,
      o2f, nullptr, NF, NF, H, W, H, W, 1);
  {
    long n = (long)B * NF * HW;
    add_inplace_k<<<dim3((unsigned)((n + 255) / 256)), dim3(256), 0, stream>>>(outL1, o2f, n);
  }
}
